// TransformerEncoder_70171175682514
// MI455X (gfx1250) — compile-verified, matrix operand order checked by reading, not run on hardware
//
#include <hip/hip_runtime.h>
#include <hip/hip_bf16.h>

// ---------------- CDNA5 WMMA types ----------------
typedef __attribute__((ext_vector_type(16))) __bf16 v16bf;
typedef __attribute__((ext_vector_type(8)))  float  v8f;

union BF16Frag {
    v16bf v;
    unsigned short s[16];
    uint4 q[2];
};

__device__ __forceinline__ unsigned short f2bf(float f) {
    unsigned int u = __float_as_uint(f);
    unsigned int r = u + 0x7FFFu + ((u >> 16) & 1u);
    return (unsigned short)(r >> 16);
}

// ---------------- f32 -> bf16 conversion ----------------
__global__ void k_f32_to_bf16(const float* __restrict__ src,
                              unsigned short* __restrict__ dst, long n) {
    long i = (long)blockIdx.x * blockDim.x + threadIdx.x;
    long stride = (long)gridDim.x * blockDim.x;
    for (; i < n; i += stride) dst[i] = f2bf(src[i]);
}

// ---------------- GEMM: C[M,N] = A[M,K] * W[N,K]^T + bias ----------------
// 128x128 tile, 8 waves (2x4), double-buffered LDS fed by
// GLOBAL_LOAD_ASYNC_TO_LDS_B128 (ASYNCcnt), 8 WMMAs per wave per 32-K step.
// MODE 0: f32 out        MODE 1: bf16 out      MODE 2: GELU -> bf16
// MODE 3: f32 out + bf16 out2                  MODE 4: bf16 V-transpose [B,H,64,T]
#define BM 128
#define BN 128
#define BK 32
#define LDSS 40   // padded row stride (ushorts) -> conflict-free b128 LDS reads

template<int MODE>
__global__ void k_gemm(const unsigned short* __restrict__ A,
                       const unsigned short* __restrict__ W,
                       const float* __restrict__ bias,
                       void* __restrict__ out, void* __restrict__ out2,
                       int M, int N, int K) {
    __shared__ __align__(16) unsigned short As[2][BM * LDSS];
    __shared__ __align__(16) unsigned short Ws[2][BN * LDSS];
    const int tid  = threadIdx.x;
    const int lane = tid & 31;
    const int wave = tid >> 5;
    const int r16  = lane & 15;
    const int hi   = lane >> 4;
    const int wm   = wave >> 2;          // 0..1 : 64-row strip
    const int wn   = wave & 3;           // 0..3 : 32-col strip
    const int mbase = blockIdx.y * BM;
    const int nbase = blockIdx.x * BN;

    v8f acc[4][2] = {};

    const int ldrow = tid >> 1;          // 0..127
    const int ldcol = (tid & 1) * 16;    // 0 or 16

    const unsigned short* gA = A + (size_t)(mbase + ldrow) * K + ldcol;
    const unsigned short* gW = W + (size_t)(nbase + ldrow) * K + ldcol;
    const unsigned lA = (unsigned)(size_t)&As[0][ldrow * LDSS + ldcol];
    const unsigned lW = (unsigned)(size_t)&Ws[0][ldrow * LDSS + ldcol];
    const unsigned bufA = (unsigned)(BM * LDSS * sizeof(unsigned short));
    const unsigned bufW = (unsigned)(BN * LDSS * sizeof(unsigned short));

    auto issue_tile = [&](int buf, int kk) {
        unsigned ao = lA + (unsigned)buf * bufA;
        unsigned wo = lW + (unsigned)buf * bufW;
        const unsigned short* pa = gA + kk;
        const unsigned short* pw = gW + kk;
        asm volatile("global_load_async_to_lds_b128 %0, %1, off"
                     :: "v"(ao), "v"(pa) : "memory");
        asm volatile("global_load_async_to_lds_b128 %0, %1, off"
                     :: "v"(ao + 16u), "v"(pa + 8) : "memory");
        asm volatile("global_load_async_to_lds_b128 %0, %1, off"
                     :: "v"(wo), "v"(pw) : "memory");
        asm volatile("global_load_async_to_lds_b128 %0, %1, off"
                     :: "v"(wo + 16u), "v"(pw + 8) : "memory");
    };

    issue_tile(0, 0);
    asm volatile("s_wait_asynccnt 0x0" ::: "memory");
    __syncthreads();

    const int nsteps = K / BK;
    for (int s = 0; s < nsteps; s++) {
        const int cur = s & 1;
        if (s + 1 < nsteps) issue_tile(cur ^ 1, (s + 1) * BK);

        const unsigned short* As_c = &As[cur][0];
        const unsigned short* Ws_c = &Ws[cur][0];
        BF16Frag af[4], bfr[2];
        #pragma unroll
        for (int mt = 0; mt < 4; mt++) {
            const unsigned short* ap = As_c + (wm * 64 + mt * 16 + r16) * LDSS;
            af[mt].q[0] = *(const uint4*)(ap + hi * 8);
            af[mt].q[1] = *(const uint4*)(ap + 16 + hi * 8);
        }
        #pragma unroll
        for (int nt = 0; nt < 2; nt++) {
            const unsigned short* wp = Ws_c + (wn * 32 + nt * 16 + r16) * LDSS;
            bfr[nt].q[0] = *(const uint4*)(wp + hi * 8);
            bfr[nt].q[1] = *(const uint4*)(wp + 16 + hi * 8);
        }
        #pragma unroll
        for (int mt = 0; mt < 4; mt++)
            #pragma unroll
            for (int nt = 0; nt < 2; nt++)
                acc[mt][nt] = __builtin_amdgcn_wmma_f32_16x16x32_bf16(
                    false, af[mt].v, false, bfr[nt].v, (short)0, acc[mt][nt], false, false);

        asm volatile("s_wait_asynccnt 0x0" ::: "memory");
        __syncthreads();
    }

    #pragma unroll
    for (int mt = 0; mt < 4; mt++) {
        #pragma unroll
        for (int nt = 0; nt < 2; nt++) {
            int n = nbase + wn * 32 + nt * 16 + r16;
            float bv = bias ? bias[n] : 0.f;
            #pragma unroll
            for (int r = 0; r < 8; r++) {
                int m = mbase + wm * 64 + mt * 16 + r + hi * 8;
                float v = acc[mt][nt][r] + bv;
                if (MODE == 0) {
                    ((float*)out)[(size_t)m * N + n] = v;
                } else if (MODE == 1) {
                    ((unsigned short*)out)[(size_t)m * N + n] = f2bf(v);
                } else if (MODE == 2) {
                    float g = 0.5f * v * (1.0f + erff(v * 0.70710678118f));
                    ((unsigned short*)out)[(size_t)m * N + n] = f2bf(g);
                } else if (MODE == 3) {
                    ((float*)out)[(size_t)m * N + n] = v;
                    ((unsigned short*)out2)[(size_t)m * N + n] = f2bf(v);
                } else { // MODE 4: V stored transposed: [B,H,64,T], m=b*512+t, n=h*64+d
                    int b = m >> 9, t = m & 511;
                    int hh = n >> 6, d = n & 63;
                    ((unsigned short*)out)[((((size_t)b * 16 + hh) * 64 + d) << 9) + t] = f2bf(v);
                }
            }
        }
    }
}

// ---------------- Flash attention with ALiBi ----------------
// qk: bf16 [B,T,2048] (Q cols 0..1023, K cols 1024..2047); vt: bf16 [B,H,64,T]
// out: bf16 [B,T,1024]. Grid (T/64, B*H), block 128 (4 waves, 16 q-rows/wave).
// Computes S^T = K*Q^T so softmax stats are per-lane scalars and exp(S^T)
// lands directly in B-fragment layout for O^T = V^T * P^T.
__global__ void k_attention(const unsigned short* __restrict__ qk,
                            const unsigned short* __restrict__ vt,
                            unsigned short* __restrict__ out) {
    const int bh   = blockIdx.y;
    const int b    = bh >> 4;
    const int h    = bh & 15;
    const int wave = threadIdx.x >> 5;
    const int lane = threadIdx.x & 31;
    const int r16  = lane & 15;
    const int hi   = lane >> 4;
    const int t0   = blockIdx.x * 64 + wave * 16;
    const int qrow = t0 + r16;                       // this lane's query index
    const float slope = exp2f(-0.5f * (float)(h + 1));
    const float scale = 0.125f;                      // 1/sqrt(64)

    BF16Frag bq0, bq1;
    {
        const unsigned short* qp = qk + ((size_t)(b * 512 + qrow) * 2048) + h * 64;
        bq0.q[0] = *(const uint4*)(qp + hi * 8);
        bq0.q[1] = *(const uint4*)(qp + 16 + hi * 8);
        bq1.q[0] = *(const uint4*)(qp + 32 + hi * 8);
        bq1.q[1] = *(const uint4*)(qp + 48 + hi * 8);
    }

    v8f o[4] = {};
    float mrun = -1e30f, srun = 0.f;

    for (int kb = 0; kb < 512; kb += 32) {
        const unsigned short* kpa = qk + ((size_t)(b * 512 + kb + r16) * 2048) + 1024 + h * 64;
        const unsigned short* kpb = kpa + (size_t)16 * 2048;
        BF16Frag ka0, ka1, kc0, kc1;
        ka0.q[0] = *(const uint4*)(kpa + hi * 8);      ka0.q[1] = *(const uint4*)(kpa + 16 + hi * 8);
        ka1.q[0] = *(const uint4*)(kpa + 32 + hi * 8); ka1.q[1] = *(const uint4*)(kpa + 48 + hi * 8);
        kc0.q[0] = *(const uint4*)(kpb + hi * 8);      kc0.q[1] = *(const uint4*)(kpb + 16 + hi * 8);
        kc1.q[0] = *(const uint4*)(kpb + 32 + hi * 8); kc1.q[1] = *(const uint4*)(kpb + 48 + hi * 8);

        v8f sa = {}, sb = {};
        sa = __builtin_amdgcn_wmma_f32_16x16x32_bf16(false, ka0.v, false, bq0.v, (short)0, sa, false, false);
        sa = __builtin_amdgcn_wmma_f32_16x16x32_bf16(false, ka1.v, false, bq1.v, (short)0, sa, false, false);
        sb = __builtin_amdgcn_wmma_f32_16x16x32_bf16(false, kc0.v, false, bq0.v, (short)0, sb, false, false);
        sb = __builtin_amdgcn_wmma_f32_16x16x32_bf16(false, kc1.v, false, bq1.v, (short)0, sb, false, false);

        float va[8], vb[8];
        float lmax = -1e30f;
        #pragma unroll
        for (int r = 0; r < 8; r++) {
            int ja = kb + r + hi * 8;
            int jb = kb + 16 + r + hi * 8;
            va[r] = sa[r] * scale - slope * fabsf((float)(qrow - ja));
            vb[r] = sb[r] * scale - slope * fabsf((float)(qrow - jb));
            lmax = fmaxf(lmax, fmaxf(va[r], vb[r]));
        }
        lmax = fmaxf(lmax, __shfl_xor(lmax, 16, 32));
        float mnew = fmaxf(mrun, lmax);
        float c = __expf(mrun - mnew);
        float lsum = 0.f;
        BF16Frag pf;   // exp(S^T) packed directly into B-fragment layout
        #pragma unroll
        for (int r = 0; r < 8; r++) {
            float pa = __expf(va[r] - mnew);
            float pb = __expf(vb[r] - mnew);
            lsum += pa + pb;
            pf.s[r]     = f2bf(pa);
            pf.s[8 + r] = f2bf(pb);
        }
        lsum += __shfl_xor(lsum, 16, 32);
        srun = srun * c + lsum;
        mrun = mnew;
        #pragma unroll
        for (int dt = 0; dt < 4; dt++)
            #pragma unroll
            for (int r = 0; r < 8; r++) o[dt][r] *= c;

        const unsigned short* vbase = vt + (((size_t)(b * 16 + h) * 64) << 9) + kb;
        #pragma unroll
        for (int dt = 0; dt < 4; dt++) {
            const unsigned short* vp = vbase + (((size_t)(dt * 16 + r16)) << 9);
            BF16Frag vf;
            vf.q[0] = *(const uint4*)(vp + hi * 8);
            vf.q[1] = *(const uint4*)(vp + 16 + hi * 8);
            o[dt] = __builtin_amdgcn_wmma_f32_16x16x32_bf16(
                false, vf.v, false, pf.v, (short)0, o[dt], false, false);
        }
    }

    float inv = 1.0f / srun;
    #pragma unroll
    for (int dt = 0; dt < 4; dt++)
        #pragma unroll
        for (int r = 0; r < 8; r++) {
            int d = dt * 16 + r + hi * 8;
            out[((size_t)(b * 512 + qrow)) * 1024 + h * 64 + d] = f2bf(o[dt][r] * inv);
        }
}

// ---------------- residual + LayerNorm (row of 1024, 256 threads) ----------------
__global__ void k_ln_residual(const float* __restrict__ hin, const float* __restrict__ tin,
                              const float* __restrict__ w, const float* __restrict__ bb,
                              float* __restrict__ hout, unsigned short* __restrict__ hbf) {
    __shared__ float red[256];
    const size_t row = blockIdx.x;
    const float* hp = hin + row * 1024;
    const float* tp = tin + row * 1024;
    float v[4];
    float s = 0.f;
    #pragma unroll
    for (int i = 0; i < 4; i++) {
        int idx = threadIdx.x + i * 256;
        v[i] = hp[idx] + tp[idx];
        s += v[i];
    }
    red[threadIdx.x] = s; __syncthreads();
    for (int off = 128; off > 0; off >>= 1) {
        if (threadIdx.x < off) red[threadIdx.x] += red[threadIdx.x + off];
        __syncthreads();
    }
    float mean = red[0] * (1.0f / 1024.0f);
    __syncthreads();
    float s2 = 0.f;
    #pragma unroll
    for (int i = 0; i < 4; i++) { float d = v[i] - mean; s2 += d * d; }
    red[threadIdx.x] = s2; __syncthreads();
    for (int off = 128; off > 0; off >>= 1) {
        if (threadIdx.x < off) red[threadIdx.x] += red[threadIdx.x + off];
        __syncthreads();
    }
    float rstd = rsqrtf(red[0] * (1.0f / 1024.0f) + 1e-5f);
    #pragma unroll
    for (int i = 0; i < 4; i++) {
        int idx = threadIdx.x + i * 256;
        float o = (v[i] - mean) * rstd * w[idx] + bb[idx];
        hout[row * 1024 + idx] = o;
        hbf[row * 1024 + idx] = f2bf(o);
    }
}

// ---------------- host orchestration ----------------
extern "C" void kernel_launch(void* const* d_in, const int* in_sizes, int n_in,
                              void* d_out, int out_size, void* d_ws, size_t ws_size,
                              hipStream_t stream) {
    const float* x     = (const float*)d_in[0];
    const float* bn_w  = (const float*)d_in[1];
    const float* bn_b  = (const float*)d_in[2];
    const float* qkv_w = (const float*)d_in[3];
    const float* qkv_b = (const float*)d_in[4];
    const float* out_w = (const float*)d_in[5];
    const float* out_b = (const float*)d_in[6];
    const float* ln1_w = (const float*)d_in[7];
    const float* ln1_b = (const float*)d_in[8];
    const float* ln2_w = (const float*)d_in[9];
    const float* ln2_b = (const float*)d_in[10];
    const float* ff1_w = (const float*)d_in[11];
    const float* ff1_b = (const float*)d_in[12];
    const float* ff2_w = (const float*)d_in[13];
    const float* ff2_b = (const float*)d_in[14];
    (void)in_sizes; (void)n_in; (void)out_size; (void)ws_size;

    const int MT = 16 * 512;   // 8192 rows
    size_t off = 0;
    auto carve = [&](size_t bytes) -> char* {
        char* p = (char*)d_ws + off;
        off += (bytes + 255) & ~(size_t)255;
        return p;
    };
    unsigned short* wbn  = (unsigned short*)carve((size_t)1024 * 1024 * 2);
    unsigned short* wqkv = (unsigned short*)carve((size_t)6 * 3072 * 1024 * 2);
    unsigned short* wout = (unsigned short*)carve((size_t)6 * 1024 * 1024 * 2);
    unsigned short* wff1 = (unsigned short*)carve((size_t)6 * 1024 * 1024 * 2);
    unsigned short* wff2 = (unsigned short*)carve((size_t)6 * 1024 * 1024 * 2);
    unsigned short* xbf  = (unsigned short*)carve((size_t)MT * 1024 * 2);
    float*          hF   = (float*)         carve((size_t)MT * 1024 * 4);
    unsigned short* hB   = (unsigned short*)carve((size_t)MT * 1024 * 2);
    unsigned short* qkb  = (unsigned short*)carve((size_t)MT * 2048 * 2);
    unsigned short* vtb  = (unsigned short*)carve((size_t)MT * 1024 * 2);
    unsigned short* atb  = (unsigned short*)carve((size_t)MT * 1024 * 2);
    float*          tmpF = (float*)         carve((size_t)MT * 1024 * 4);
    unsigned short* fbuf = (unsigned short*)carve((size_t)MT * 1024 * 2);

    // weight + input conversion to bf16 (every call; no cached state)
    k_f32_to_bf16<<<2048, 256, 0, stream>>>(bn_w,  wbn,  (long)1024 * 1024);
    k_f32_to_bf16<<<4096, 256, 0, stream>>>(qkv_w, wqkv, (long)6 * 3072 * 1024);
    k_f32_to_bf16<<<2048, 256, 0, stream>>>(out_w, wout, (long)6 * 1024 * 1024);
    k_f32_to_bf16<<<2048, 256, 0, stream>>>(ff1_w, wff1, (long)6 * 1024 * 1024);
    k_f32_to_bf16<<<2048, 256, 0, stream>>>(ff2_w, wff2, (long)6 * 1024 * 1024);
    k_f32_to_bf16<<<2048, 256, 0, stream>>>(x,     xbf,  (long)MT * 1024);

    dim3 blk(256);
    dim3 g1024(1024 / BN, MT / BM);
    dim3 g2048(2048 / BN, MT / BM);

    // bottleneck: h = x @ bn_w^T + bn_b  (f32 state + bf16 copy)
    k_gemm<3><<<g1024, blk, 0, stream>>>(xbf, wbn, bn_b, hF, hB, MT, 1024, 1024);

    for (int l = 0; l < 6; l++) {
        const unsigned short* wq = wqkv + (size_t)l * 3072 * 1024;
        // Q,K -> [B,T,2048]
        k_gemm<1><<<g2048, blk, 0, stream>>>(hB, wq, qkv_b + (size_t)l * 3072,
                                             qkb, nullptr, MT, 2048, 1024);
        // V -> transposed [B,H,64,T]
        k_gemm<4><<<g1024, blk, 0, stream>>>(hB, wq + (size_t)2048 * 1024,
                                             qkv_b + (size_t)l * 3072 + 2048,
                                             vtb, nullptr, MT, 1024, 1024);
        // flash attention with ALiBi
        k_attention<<<dim3(8, 256), dim3(128), 0, stream>>>(qkb, vtb, atb);
        // out projection
        k_gemm<0><<<g1024, blk, 0, stream>>>(atb, wout + (size_t)l * 1024 * 1024,
                                             out_b + (size_t)l * 1024,
                                             tmpF, nullptr, MT, 1024, 1024);
        k_ln_residual<<<MT, 256, 0, stream>>>(hF, tmpF, ln1_w + (size_t)l * 1024,
                                              ln1_b + (size_t)l * 1024, hF, hB);
        // FF1 + exact GELU
        k_gemm<2><<<g1024, blk, 0, stream>>>(hB, wff1 + (size_t)l * 1024 * 1024,
                                             ff1_b + (size_t)l * 1024,
                                             fbuf, nullptr, MT, 1024, 1024);
        // FF2
        k_gemm<0><<<g1024, blk, 0, stream>>>(fbuf, wff2 + (size_t)l * 1024 * 1024,
                                             ff2_b + (size_t)l * 1024,
                                             tmpF, nullptr, MT, 1024, 1024);
        float* hdst = (l == 5) ? (float*)d_out : hF;
        k_ln_residual<<<MT, 256, 0, stream>>>(hF, tmpF, ln2_w + (size_t)l * 1024,
                                              ln2_b + (size_t)l * 1024, hdst, hB);
    }
}
